// SpMiddleResNetFHDv2_57801669870241
// MI455X (gfx1250) — matrix operand order repaired and checked by reading, not compiled, not run on hardware
//
#include <hip/hip_runtime.h>

// ---------------------------------------------------------------------------
// CDNA5 (gfx1250) implicit-GEMM 3D conv network using v_wmma_f32_16x16x32_bf16
// - Multi-M-tile blocking: one B (im2col) tile feeds MT weight tiles per chunk
// - Weight tiles staged to LDS via async global->LDS DMA (ASYNCcnt), overlapped
//   with the im2col address generation
// ---------------------------------------------------------------------------

typedef __attribute__((ext_vector_type(16))) __bf16 v16bf;
typedef __attribute__((ext_vector_type(8)))  float  v8f;

static __device__ __forceinline__ unsigned short f2bf(float f) {
  // round-to-nearest-even fp32 -> bf16
  unsigned int u = __float_as_uint(f);
  u += 0x7FFFu + ((u >> 16) & 1u);
  return (unsigned short)(u >> 16);
}

// ---------------------------------------------------------------------------
// utility kernels
// ---------------------------------------------------------------------------

__global__ void fill_zero_kernel(float* __restrict__ p, long n) {
  long i = (long)blockIdx.x * blockDim.x + threadIdx.x;
  if (i < n) p[i] = 0.0f;
}

__global__ void scatter_kernel(const float* __restrict__ vf, const int* __restrict__ coors,
                               float* __restrict__ x0, float* __restrict__ m1,
                               int N, int C, int D, int H, int W) {
  int i = blockIdx.x * blockDim.x + threadIdx.x;
  if (i >= N) return;
  int d = coors[i * 4 + 1];
  int h = coors[i * 4 + 2];
  int w = coors[i * 4 + 3];
  long sp  = ((long)d * H + h) * W + w;
  long vol = (long)D * H * W;
  for (int c = 0; c < C; ++c) x0[c * vol + sp] = vf[(long)i * C + c];
  m1[sp] = 1.0f;
}

__global__ void down_mask_kernel(const float* __restrict__ mi, float* __restrict__ mo,
                                 int Din, int Hin, int Win, int Dout, int Hout, int Wout,
                                 int kd, int kh, int kw, int sd, int sh, int sw,
                                 int pd, int ph, int pw) {
  int n = blockIdx.x * blockDim.x + threadIdx.x;
  int Nsp = Dout * Hout * Wout;
  if (n >= Nsp) return;
  int x = n % Wout;
  int t = n / Wout;
  int y = t % Hout;
  int z = t / Hout;
  float a = 0.0f;
  for (int td = 0; td < kd; ++td)
    for (int th = 0; th < kh; ++th)
      for (int tw = 0; tw < kw; ++tw) {
        int zi = z * sd - pd + td;
        int yi = y * sh - ph + th;
        int xi = x * sw - pw + tw;
        if ((unsigned)zi < (unsigned)Din && (unsigned)yi < (unsigned)Hin &&
            (unsigned)xi < (unsigned)Win)
          a += mi[((long)zi * Hin + yi) * Win + xi];
      }
  mo[n] = (a > 0.0f) ? 1.0f : 0.0f;
}

// Pack fp32 weights [Cout][Cin][taps] into per-lane WMMA A-fragments:
// out[((cb*kch + c)*32 + lane)*16 + j], matching the ISA 16-bit A-matrix 16x32
// layout: k = c*32 + (i/4)*16 + (lane/16)*8 + (i%4)*2 + e  with i=j/2, e=j%2.
__global__ void pack_kernel(const float* __restrict__ w, unsigned short* __restrict__ out,
                            int Cin, int taps, int kch, long total) {
  long idx = (long)blockIdx.x * blockDim.x + threadIdx.x;
  if (idx >= total) return;
  int  j    = (int)(idx & 15);
  int  lane = (int)((idx >> 4) & 31);
  long t    = idx >> 9;
  int  c    = (int)(t % kch);
  int  cb   = (int)(t / kch);
  int  m    = lane & 15;
  int  half = lane >> 4;
  int  i    = j >> 1;
  int  e    = j & 1;
  int  k    = c * 32 + ((i >> 2) << 4) + (half << 3) + ((i & 3) << 1) + e;
  float v = 0.0f;
  if (k < Cin * taps) {
    int tap = k / Cin;
    int ci  = k - tap * Cin;
    int co  = (cb << 4) + m;
    v = w[((long)co * Cin + ci) * taps + tap];
  }
  out[idx] = f2bf(v);
}

// ---------------------------------------------------------------------------
// implicit-GEMM conv:  out[co][n] = sum_k  W[co][k] * patch[k][n]
// K order: k = tap*Cin + ci.
// One workgroup = 4 waves = (16*MT Cout) x 64(N) tile.  Per K-chunk:
//   1) issue async global->LDS DMA of the MT weight A-tiles (ASYNCcnt)
//   2) build shared B (im2col) tile in LDS (overlaps the DMA)
//   3) s_wait_asynccnt 0 + barrier
//   4) each wave: 1 B fragment + MT A fragments (ds_load_b128) -> MT WMMAs
// Epilogue modes:
//   0: relu(acc*s + sh) * mask                       (layer_fwd)
//   1: relu((acc+b)*s + sh) * mask                   (block conv1)
//   2: relu(((acc+b)*s + sh)*mask + residual)        (block conv2)
// ---------------------------------------------------------------------------
template <int CIN, int KD, int KH, int KW, int MT>
__global__ __launch_bounds__(128) void conv_wmma_kernel(
    const unsigned short* __restrict__ wpk,
    const float* __restrict__ in, float* __restrict__ out,
    const float* __restrict__ scale, const float* __restrict__ shift,
    const float* __restrict__ bias, const float* __restrict__ mask,
    const float* __restrict__ res,
    int Din, int Hin, int Win, int Dout, int Hout, int Wout,
    int strD, int strH, int strW, int padD, int padH, int padW,
    int kchunks, int mode) {
  constexpr int TAPS = KD * KH * KW;
  const int Nsp    = Dout * Hout * Wout;
  const int tid    = threadIdx.x;
  const int lane   = tid & 31;
  const int wave   = tid >> 5;
  const int n_base = blockIdx.x * 64;
  const int cb0    = blockIdx.y * MT;  // first 16-channel M tile of this block

  __shared__ unsigned short sB[64][34];           // 64 N-rows x 32 K (bf16), +2 pad
  __shared__ alignas(32) unsigned short sA[MT * 512];  // MT x (32 lanes x 16 bf16)

  v8f acc[MT];
#pragma unroll
  for (int mt = 0; mt < MT; ++mt) acc[mt] = (v8f){0.f, 0.f, 0.f, 0.f, 0.f, 0.f, 0.f, 0.f};

  // ---- chunk-invariant spatial decomposition for this thread's 16 rows ----
  int ix0[16], iy0[16], iz0[16], rb[16];
#pragma unroll
  for (int it = 0; it < 16; ++it) {
    int n = n_base + wave + (it << 2);
    if (n < Nsp) {
      int x = n % Wout;
      int t = n / Wout;
      int y = t % Hout;
      int z = t / Hout;
      ix0[it] = x * strW - padW;
      iy0[it] = y * strH - padH;
      iz0[it] = z * strD - padD;
      rb[it]  = (iz0[it] * Hin + iy0[it]) * Win + ix0[it];
    } else {
      ix0[it] = -0x40000000;  // forces every bounds check to fail
      iy0[it] = -0x40000000;
      iz0[it] = -0x40000000;
      rb[it]  = 0;
    }
  }

  const long chvol = (long)Din * Hin * Win;

  for (int c = 0; c < kchunks; ++c) {
    // -------- 1) async DMA: MT KB of packed weights -> LDS (ASYNCcnt) --------
    // Each selected lane moves 16B; the whole chunk tile is fetched exactly
    // once per workgroup instead of once per wave.
#pragma unroll
    for (int e = tid; e < MT * 64; e += 128) {
      const int mt   = e >> 6;
      const int part = e & 63;
      const unsigned short* g =
          wpk + ((long)(cb0 + mt) * kchunks + c) * 512 + part * 8;
      unsigned ldsa = (unsigned)(unsigned long long)&sA[mt * 512 + part * 8];
      asm volatile("global_load_async_to_lds_b128 %0, %1, off"
                   :: "v"(ldsa), "v"(g)
                   : "memory");
    }

    // -------- 2) cooperative im2col gather into LDS (overlaps the DMA) ------
    // thread (wave, lane) fills column k=lane of rows {wave, wave+4, ...}
    const int k   = c * 32 + lane;
    const int tap = k / CIN;  // CIN is compile-time: shifts / magic mul
    const int ci  = k - tap * CIN;
    const bool kvalid = (tap < TAPS);
    int td = 0, th = 0, tw = 0;
    if (kvalid) {
      tw = tap % KW;
      int t2 = tap / KW;
      th = t2 % KH;
      td = t2 / KH;
    }
    const float* inc = in + (long)ci * chvol;
    const int    off = (td * Hin + th) * Win + tw;

#pragma unroll
    for (int it = 0; it < 16; ++it) {
      int xi = ix0[it] + tw;
      int yi = iy0[it] + th;
      int zi = iz0[it] + td;
      float v = 0.0f;
      if (kvalid && (unsigned)xi < (unsigned)Win && (unsigned)yi < (unsigned)Hin &&
          (unsigned)zi < (unsigned)Din)
        v = inc[rb[it] + off];
      sB[wave + (it << 2)][lane] = f2bf(v);
    }

    // -------- 3) wait for this wave's async DMA, then workgroup barrier -----
    asm volatile("s_wait_asynccnt 0x0" ::: "memory");
    __syncthreads();

    // -------- 4) B fragment: n = lane%16, k-half selected by lane/16 --------
    union { unsigned int u[8]; v16bf v; } bf;
    const int ncol = (wave << 4) + (lane & 15);
    const unsigned int* row = reinterpret_cast<const unsigned int*>(&sB[ncol][0]);
    const int koff = (lane >> 4) << 2;  // dword offset 0 or 4 (K=0..7,16..23 / 8..15,24..31)
#pragma unroll
    for (int i = 0; i < 8; ++i) bf.u[i] = row[koff + (i & 3) + ((i >> 2) << 3)];

    // -------- MT weight fragments (from LDS) x 1 B fragment -> MT WMMAs -----
#pragma unroll
    for (int mt = 0; mt < MT; ++mt) {
      const v16bf afrag =
          *reinterpret_cast<const v16bf*>(&sA[mt * 512 + lane * 16]);
      acc[mt] = __builtin_amdgcn_wmma_f32_16x16x32_bf16(
          /*neg_a=*/false, afrag, /*neg_b=*/false, bf.v,
          /*c_mod=*/(short)0, acc[mt], /*reuse_a=*/false, /*reuse_b=*/false);
    }
    __syncthreads();
  }

  // -------- epilogue: C/D layout  M = vgpr + 8*(lane>=16), N = lane%16 --------
  const int nn = n_base + (wave << 4) + (lane & 15);
  if (nn < Nsp) {
    const float mval = mask[nn];
    const int   mhi  = (lane >> 4) << 3;
#pragma unroll
    for (int mt = 0; mt < MT; ++mt) {
#pragma unroll
      for (int j = 0; j < 8; ++j) {
        int co  = ((cb0 + mt) << 4) + mhi + j;
        float v = acc[mt][j];
        if (mode != 0) v += bias[co];
        v = v * scale[co] + shift[co];
        if (mode == 2) {
          v = v * mval + res[(long)co * Nsp + nn];
          v = (v > 0.f) ? v : 0.f;
        } else {
          v = ((v > 0.f) ? v : 0.f) * mval;
        }
        out[(long)co * Nsp + nn] = v;
      }
    }
  }
}

// ---------------------------------------------------------------------------
// host-side launch helpers
// ---------------------------------------------------------------------------

static void conv_launch(hipStream_t st, const unsigned short* wpk,
                        const float* in, float* out,
                        const float* scale, const float* shift, const float* bias,
                        const float* mask, const float* res,
                        int Cin, int Cout,
                        int Din, int Hin, int Win, int Dout, int Hout, int Wout,
                        int kd, int kh, int kw, int sd, int sh, int sw,
                        int pd, int ph, int pw, int mode) {
  int taps = kd * kh * kw;
  int kch  = (Cin * taps + 31) / 32;
  int Nsp  = Dout * Hout * Wout;
  int mt   = Cout / 16;
  if (mt > 4) mt = 4;
  dim3 grid((unsigned)((Nsp + 63) / 64), (unsigned)(Cout / (16 * mt))), block(128);
#define CW(CI, KD_, KH_, KW_, MT_)                                                \
  conv_wmma_kernel<CI, KD_, KH_, KW_, MT_><<<grid, block, 0, st>>>(               \
      wpk, in, out, scale, shift, bias, mask, res, Din, Hin, Win, Dout, Hout,     \
      Wout, sd, sh, sw, pd, ph, pw, kch, mode)
  if (Cin == 5)                    CW(5, 3, 3, 3, 1);
  else if (Cin == 16 && mt == 1)   CW(16, 3, 3, 3, 1);
  else if (Cin == 16)              CW(16, 3, 3, 3, 2);
  else if (Cin == 32 && mt == 2)   CW(32, 3, 3, 3, 2);
  else if (Cin == 32)              CW(32, 3, 3, 3, 4);
  else if (Cin == 64)              CW(64, 3, 3, 3, 4);
  else if (kw == 3)                CW(128, 3, 3, 3, 4);
  else                             CW(128, 3, 1, 1, 4);
#undef CW
}

// d_in leaf indices (JAX pytree flatten: top-level insertion order, nested dicts
// sorted by key, lists in order):
//   0 voxel_features
//   1..3     extra: scale, shift, w
//   4..27    stage1_blocks[0..2]: (b1,b2,h1,h2,s1,s2,w1,w2) each
//   28..51   stage2.blocks[0..2]
//   52..54   stage2.down (scale,shift,w);  55..57 stage2.trans
//   58..81   stage3.blocks; 82..84 down; 85..87 trans
//   88..111  stage4.blocks; 112..114 down; 115..117 trans
//   118..120 stem[0]; 121..123 stem[1]
//   124 coors, 125 batch_size, 126.. input_shape
extern "C" void kernel_launch(void* const* d_in, const int* in_sizes, int n_in,
                              void* d_out, int out_size, void* d_ws, size_t ws_size,
                              hipStream_t stream) {
  (void)in_sizes; (void)n_in; (void)out_size; (void)ws_size;
  auto F = [&](int i) { return (const float*)d_in[i]; };
  const int* coors = (const int*)d_in[124];

  const int NVOX = 60000, CIN0 = 5;
  const int D1 = 41, H1 = 160, W1 = 160, N1 = D1 * H1 * W1;
  const int D2 = 21, H2 = 80,  W2 = 80,  N2 = D2 * H2 * W2;
  const int D3 = 11, H3 = 40,  W3 = 40,  N3 = D3 * H3 * W3;
  const int D4 = 5,  H4 = 20,  W4 = 20,  N4 = D4 * H4 * W4;
  const int D5 = 2,  H5 = 20,  W5 = 20,  N5 = D5 * H5 * W5;

  // bump allocator over workspace
  size_t off = 0;
  char*  wsb = (char*)d_ws;
  auto alloc = [&](size_t bytes) -> void* {
    void* p = wsb + off;
    off += (bytes + 255) & ~(size_t)255;
    return p;
  };

  float* x0   = (float*)alloc((size_t)CIN0 * N1 * 4);
  float* bufA = (float*)alloc((size_t)16 * N1 * 4);
  float* bufB = (float*)alloc((size_t)16 * N1 * 4);
  float* m1 = (float*)alloc((size_t)N1 * 4);
  float* m2 = (float*)alloc((size_t)N2 * 4);
  float* m3 = (float*)alloc((size_t)N3 * 4);
  float* m4 = (float*)alloc((size_t)N4 * 4);
  float* m5 = (float*)alloc((size_t)N5 * 4);

  // ---- pack all conv weights to bf16 WMMA A-fragment layout ----
  auto packw = [&](int widx, int Cout, int Cin, int taps) -> unsigned short* {
    int  kch   = (Cin * taps + 31) / 32;
    long total = (long)(Cout / 16) * kch * 512;
    unsigned short* p = (unsigned short*)alloc((size_t)total * 2);
    pack_kernel<<<dim3((unsigned)((total + 255) / 256)), dim3(256), 0, stream>>>(
        F(widx), p, Cin, taps, kch, total);
    return p;
  };

  const int blkBase[4] = {4, 28, 58, 88};
  const int blkC[4]    = {16, 32, 64, 128};
  unsigned short* wp_stem0 = packw(120, 16, 5, 27);
  unsigned short* wp_stem1 = packw(123, 16, 16, 27);
  unsigned short* wp_b[12][2];
  for (int s = 0; s < 4; ++s)
    for (int b = 0; b < 3; ++b) {
      int base = blkBase[s] + 8 * b;
      wp_b[s * 3 + b][0] = packw(base + 6, blkC[s], blkC[s], 27);
      wp_b[s * 3 + b][1] = packw(base + 7, blkC[s], blkC[s], 27);
    }
  unsigned short* wp_down[3]  = {packw(54, 32, 16, 27), packw(84, 64, 32, 27),
                                 packw(114, 128, 64, 27)};
  unsigned short* wp_trans[3] = {packw(57, 32, 32, 27), packw(87, 64, 64, 27),
                                 packw(117, 128, 128, 27)};
  unsigned short* wp_extra = packw(3, 128, 128, 3);

  // ---- scatter voxels into dense grid + mask ----
  long zc = (long)CIN0 * N1;
  fill_zero_kernel<<<dim3((unsigned)((zc + 255) / 256)), dim3(256), 0, stream>>>(x0, zc);
  fill_zero_kernel<<<dim3((unsigned)((N1 + 255) / 256)), dim3(256), 0, stream>>>(m1, N1);
  scatter_kernel<<<dim3((NVOX + 255) / 256), dim3(256), 0, stream>>>(
      F(0), coors, x0, m1, NVOX, CIN0, D1, H1, W1);

  // ---- output regions double as stage activations ----
  float* outp   = (float*)d_out;
  float* ret    = outp;                        // (1, 256, 20, 20)
  float* feats0 = outp + 102400;               // (16, 41,160,160)
  float* feats1 = feats0 + (size_t)16 * N1;    // (32, 21, 80, 80)
  float* feats2 = feats1 + (size_t)32 * N2;    // (64, 11, 40, 40)
  float* feats3 = feats2 + (size_t)64 * N3;    // (128, 5, 20, 20)

  // ---- stem ----
  conv_launch(stream, wp_stem0, x0, bufA, F(118), F(119), nullptr, m1, nullptr,
              5, 16, D1, H1, W1, D1, H1, W1, 3, 3, 3, 1, 1, 1, 1, 1, 1, 0);
  conv_launch(stream, wp_stem1, bufA, bufB, F(121), F(122), nullptr, m1, nullptr,
              16, 16, D1, H1, W1, D1, H1, W1, 3, 3, 3, 1, 1, 1, 1, 1, 1, 0);

  auto run_block = [&](int base, unsigned short* w1, unsigned short* w2, int C,
                       float* X, float* T, float* XO, const float* mk,
                       int D, int H, int W) {
    conv_launch(stream, w1, X, T, F(base + 4), F(base + 2), F(base + 0), mk, nullptr,
                C, C, D, H, W, D, H, W, 3, 3, 3, 1, 1, 1, 1, 1, 1, 1);
    conv_launch(stream, w2, T, XO, F(base + 5), F(base + 3), F(base + 1), mk, X,
                C, C, D, H, W, D, H, W, 3, 3, 3, 1, 1, 1, 1, 1, 1, 2);
  };

  // ---- stage 1 (X lives in bufB; last block writes feats0) ----
  run_block(4,  wp_b[0][0], wp_b[0][1], 16, bufB, bufA, bufB,   m1, D1, H1, W1);
  run_block(12, wp_b[1][0], wp_b[1][1], 16, bufB, bufA, bufB,   m1, D1, H1, W1);
  run_block(20, wp_b[2][0], wp_b[2][1], 16, bufB, bufA, feats0, m1, D1, H1, W1);

  // ---- stages 2..4 ----
  const float* fin[3]  = {feats0, feats1, feats2};
  float* fout[3]       = {feats1, feats2, feats3};
  float* mprev[3]      = {m1, m2, m3};
  float* mcur[3]       = {m2, m3, m4};
  const int Di[3] = {D1, D2, D3}, Hi[3] = {H1, H2, H3}, Wi[3] = {W1, W2, W3};
  const int Do[3] = {D2, D3, D4}, Ho[3] = {H2, H3, H4}, Wo[3] = {W2, W3, W4};
  const int Ci[3] = {16, 32, 64}, Co[3] = {32, 64, 128};
  const int padd[3]     = {1, 1, 0};
  const int downIdx[3]  = {52, 82, 112};
  const int transIdx[3] = {55, 85, 115};

  for (int s = 0; s < 3; ++s) {
    int Nsp = Do[s] * Ho[s] * Wo[s];
    down_mask_kernel<<<dim3((unsigned)((Nsp + 255) / 256)), dim3(256), 0, stream>>>(
        mprev[s], mcur[s], Di[s], Hi[s], Wi[s], Do[s], Ho[s], Wo[s],
        3, 3, 3, 2, 2, 2, padd[s], 1, 1);
    conv_launch(stream, wp_down[s], fin[s], bufA, F(downIdx[s]), F(downIdx[s] + 1),
                nullptr, mcur[s], nullptr, Ci[s], Co[s],
                Di[s], Hi[s], Wi[s], Do[s], Ho[s], Wo[s],
                3, 3, 3, 2, 2, 2, padd[s], 1, 1, 0);
    conv_launch(stream, wp_trans[s], bufA, bufB, F(transIdx[s]), F(transIdx[s] + 1),
                nullptr, mcur[s], nullptr, Co[s], Co[s],
                Do[s], Ho[s], Wo[s], Do[s], Ho[s], Wo[s],
                3, 3, 3, 1, 1, 1, 1, 1, 1, 0);
    int bb = blkBase[s + 1];
    int bi = (s + 1) * 3;
    run_block(bb,      wp_b[bi + 0][0], wp_b[bi + 0][1], Co[s], bufB, bufA, bufB,
              mcur[s], Do[s], Ho[s], Wo[s]);
    run_block(bb + 8,  wp_b[bi + 1][0], wp_b[bi + 1][1], Co[s], bufB, bufA, bufB,
              mcur[s], Do[s], Ho[s], Wo[s]);
    run_block(bb + 16, wp_b[bi + 2][0], wp_b[bi + 2][1], Co[s], bufB, bufA, fout[s],
              mcur[s], Do[s], Ho[s], Wo[s]);
  }

  // ---- extra layer: (3,1,1) conv, stride (2,1,1), pad 0 -> ret ----
  down_mask_kernel<<<dim3((unsigned)((N5 + 255) / 256)), dim3(256), 0, stream>>>(
      m4, m5, D4, H4, W4, D5, H5, W5, 3, 1, 1, 2, 1, 1, 0, 0, 0);
  conv_launch(stream, wp_extra, feats3, ret, F(1), F(2), nullptr, m5, nullptr,
              128, 128, D4, H4, W4, D5, H5, W5, 3, 1, 1, 2, 1, 1, 0, 0, 0, 0);
}
